// MPNN_41394894799199
// MI455X (gfx1250) — matrix operand order repaired and checked by reading, not compiled, not run on hardware
//
#include <hip/hip_runtime.h>
#include <hip/hip_bf16.h>

// ---------------------------------------------------------------------------
// D-MPNN on MI455X (gfx1250): bf16 WMMA GEMMs + bandwidth-bound gathers.
// ---------------------------------------------------------------------------

typedef __bf16 bf16;
typedef __attribute__((ext_vector_type(16))) __bf16 v16bf;
typedef __attribute__((ext_vector_type(8)))  __bf16 v8bf;
typedef __attribute__((ext_vector_type(8)))  float  v8f;

#define N_ATOMS 65536
#define N_BONDS 262144
#define MAX_NB  6
#define HID     256
#define ATOM_F  133
#define BOND_F  147
#define N_MOLS  2048

#define KP_BOND 160   // BOND_F padded to mult of 32
#define KP_CAT  416   // (ATOM_F + HID)=389 padded to mult of 32

#define BM 128
#define BN 64
#define BK 32

// ---------------------------------------------------------------------------
// Conversion / layout prepass kernels
// ---------------------------------------------------------------------------

// f_bonds [N_BONDS][147] f32 -> [N_BONDS][160] bf16 (zero pad)
__global__ __launch_bounds__(256)
void mpnn_conv_fbonds(const float* __restrict__ fb, bf16* __restrict__ out) {
    long long gid = (long long)blockIdx.x * 256 + threadIdx.x;
    if (gid >= (long long)N_BONDS * KP_BOND) return;
    int bond = (int)(gid / KP_BOND);
    int col  = (int)(gid % KP_BOND);
    float v = (col < BOND_F) ? fb[(size_t)bond * BOND_F + col] : 0.0f;
    out[gid] = (bf16)v;
}

// W [K][N] f32 -> Wt [N][Kpad] bf16 (transposed, zero pad rows >= K)
__global__ __launch_bounds__(256)
void mpnn_conv_weight_t(const float* __restrict__ W, int K, int N, int Kpad,
                        bf16* __restrict__ Wt) {
    int gid = blockIdx.x * 256 + threadIdx.x;
    if (gid >= N * Kpad) return;
    int n = gid / Kpad;
    int k = gid % Kpad;
    float v = (k < K) ? W[(size_t)k * N + n] : 0.0f;
    Wt[gid] = (bf16)v;
}

// ---------------------------------------------------------------------------
// Tiled WMMA GEMM: C[M][256] = A[M][lda(bf16)] x Bt[N=256][ldb(bf16)]^T
// Block tile 128x64, 8 waves (4 along M x 2 along N), wave tile 32x32.
// mode 0: out_inp = acc ; out_msg = relu(acc)                     (W_i GEMM)
// mode 1: out_msg = relu(acc + inp_in)                            (W_m GEMM)
// mode 2: h = relu(acc + bias[col]); atomicAdd(mol_sums[mol,col]) (W_a GEMM)
// ---------------------------------------------------------------------------
__global__ __launch_bounds__(256)
void mpnn_gemm_wmma(const bf16* __restrict__ A, int lda,
                    const bf16* __restrict__ Bt, int ldb,
                    int K, int mode,
                    bf16* __restrict__ out_inp,
                    bf16* __restrict__ out_msg,
                    const bf16* __restrict__ inp_in,
                    const float* __restrict__ bias,
                    const int* __restrict__ mol_index,
                    float* __restrict__ mol_sums) {
    __shared__ bf16 sA[BM * BK];   // [128][32]
    __shared__ bf16 sB[BN * BK];   // [64][32]  (row = N-col of original B)

    const int bm   = blockIdx.x;
    const int bn   = blockIdx.y;
    const int t    = threadIdx.x;
    const int wave = t >> 5;
    const int lane = t & 31;
    const int wm   = wave & 3;     // wave row (M)
    const int wn   = wave >> 2;    // wave col (N)
    const int l16  = lane & 15;
    const int half = lane >> 4;

    v8f acc[2][2];
    for (int mt = 0; mt < 2; ++mt)
        for (int nt = 0; nt < 2; ++nt)
            for (int i = 0; i < 8; ++i) acc[mt][nt][i] = 0.0f;

    for (int k0 = 0; k0 < K; k0 += BK) {
        // ---- stage A tile: 128x32 bf16, each thread copies 2x16B
        {
            int row = t >> 1;
            int col = (t & 1) * 16;
            const uint4* src = (const uint4*)(A + (size_t)(bm * BM + row) * lda + k0 + col);
            uint4* dst = (uint4*)(sA + row * BK + col);
            dst[0] = src[0];
            dst[1] = src[1];
        }
        // ---- stage B tile: 64x32 bf16, each thread copies 1x16B
        {
            int n   = t >> 2;
            int col = (t & 3) * 8;
            *(uint4*)(sB + n * BK + col) =
                *(const uint4*)(Bt + (size_t)(bn * BN + n) * ldb + k0 + col);
        }
        // prefetch next A tile row into L2 while we compute
        if (k0 + BK < K) {
            int row = t >> 1;
            __builtin_prefetch(A + (size_t)(bm * BM + row) * lda + k0 + BK, 0, 1);
        }
        __syncthreads();

        // ---- build fragments per ISA layout
        // A 16x32 bf16: lane row = l16; K = {h*8..h*8+7} U {16+h*8..16+h*8+7}
        v16bf afrag[2];
        for (int mt = 0; mt < 2; ++mt) {
            int row = wm * 32 + mt * 16 + l16;
            v8bf lo = *(const v8bf*)(sA + row * BK + half * 8);
            v8bf hi = *(const v8bf*)(sA + row * BK + half * 8 + 16);
            for (int i = 0; i < 8; ++i) { afrag[mt][i] = lo[i]; afrag[mt][8 + i] = hi[i]; }
        }
        // B 32x16 bf16: lane col = l16; K = {h*8..h*8+7} U {16+h*8..16+h*8+7}
        v16bf bfrag[2];
        for (int nt = 0; nt < 2; ++nt) {
            int ncol = wn * 32 + nt * 16 + l16;
            v8bf lo = *(const v8bf*)(sB + ncol * BK + half * 8);
            v8bf hi = *(const v8bf*)(sB + ncol * BK + half * 8 + 16);
            for (int i = 0; i < 8; ++i) { bfrag[nt][i] = lo[i]; bfrag[nt][8 + i] = hi[i]; }
        }

        for (int mt = 0; mt < 2; ++mt)
            for (int nt = 0; nt < 2; ++nt)
                acc[mt][nt] = __builtin_amdgcn_wmma_f32_16x16x32_bf16(
                    false, afrag[mt], false, bfrag[nt],
                    (short)0, acc[mt][nt], false, false);

        __syncthreads();
    }

    // ---- epilogue; C layout: VGPR r -> M = r + 8*half, N = l16
    const int rbase = half * 8;
    for (int mt = 0; mt < 2; ++mt) {
        for (int nt = 0; nt < 2; ++nt) {
            int colg = bn * BN + wn * 32 + nt * 16 + l16;   // 0..255
            for (int r = 0; r < 8; ++r) {
                int rowg = bm * BM + wm * 32 + mt * 16 + rbase + r;
                float v = acc[mt][nt][r];
                if (mode == 0) {
                    size_t idx = (size_t)rowg * HID + colg;
                    out_inp[idx] = (bf16)v;
                    out_msg[idx] = (bf16)fmaxf(v, 0.0f);
                } else if (mode == 1) {
                    size_t idx = (size_t)rowg * HID + colg;
                    float s = v + (float)inp_in[idx];
                    out_msg[idx] = (bf16)fmaxf(s, 0.0f);
                } else {
                    float h = fmaxf(v + bias[colg], 0.0f);
                    atomicAdd(&mol_sums[(size_t)mol_index[rowg] * HID + colg], h);
                }
            }
        }
    }
}

// ---------------------------------------------------------------------------
// agg[a][h] = sum_{j<6} msg[a2b[a][j]][h]   (one thread = one atom x 8 cols)
// ---------------------------------------------------------------------------
__global__ __launch_bounds__(256)
void mpnn_atom_agg(const bf16* __restrict__ msg, const int* __restrict__ a2b,
                   float* __restrict__ agg) {
    int gid  = blockIdx.x * 256 + threadIdx.x;      // N_ATOMS * 32 items
    int atom = gid >> 5;
    int col  = (gid & 31) * 8;
    float s[8];
    for (int i = 0; i < 8; ++i) s[i] = 0.0f;
    for (int j = 0; j < MAX_NB; ++j) {
        int b = a2b[atom * MAX_NB + j];
        v8bf mv = *(const v8bf*)(msg + (size_t)b * HID + col);
        for (int i = 0; i < 8; ++i) s[i] += (float)mv[i];
    }
    v8f out;
    for (int i = 0; i < 8; ++i) out[i] = s[i];
    *(v8f*)(agg + (size_t)atom * HID + col) = out;
}

// ---------------------------------------------------------------------------
// m[b][h] = agg[b2a[b]][h] - msg[b2revb[b]][h]   (bf16 out)
// ---------------------------------------------------------------------------
__global__ __launch_bounds__(256)
void mpnn_bond_m(const float* __restrict__ agg, const bf16* __restrict__ msg,
                 const int* __restrict__ b2a, const int* __restrict__ b2revb,
                 bf16* __restrict__ m16) {
    int gid  = blockIdx.x * 256 + threadIdx.x;      // N_BONDS * 32 items
    int bond = gid >> 5;
    int col  = (gid & 31) * 8;
    int sa = b2a[bond];
    int rb = b2revb[bond];
    v8f  av = *(const v8f*)(agg + (size_t)sa * HID + col);
    v8bf mv = *(const v8bf*)(msg + (size_t)rb * HID + col);
    v8bf o;
    for (int i = 0; i < 8; ++i) o[i] = (bf16)(av[i] - (float)mv[i]);
    *(v8bf*)(m16 + (size_t)bond * HID + col) = o;
}

// ---------------------------------------------------------------------------
// a_input[a] = concat(f_atoms[a] (133), agg[a] (256), zeros (27)) -> bf16[416]
// ---------------------------------------------------------------------------
__global__ __launch_bounds__(256)
void mpnn_build_ainput(const float* __restrict__ f_atoms,
                       const float* __restrict__ agg,
                       bf16* __restrict__ ain) {
    long long gid = (long long)blockIdx.x * 256 + threadIdx.x;
    if (gid >= (long long)N_ATOMS * KP_CAT) return;
    int atom = (int)(gid / KP_CAT);
    int col  = (int)(gid % KP_CAT);
    float v;
    if (col < ATOM_F)              v = f_atoms[(size_t)atom * ATOM_F + col];
    else if (col < ATOM_F + HID)   v = agg[(size_t)atom * HID + (col - ATOM_F)];
    else                           v = 0.0f;
    ain[gid] = (bf16)v;
}

// ---------------------------------------------------------------------------
// zero mol_sums + counts; count atoms per molecule; final mean
// ---------------------------------------------------------------------------
__global__ __launch_bounds__(256)
void mpnn_zero(float* __restrict__ p, int n) {
    int gid = blockIdx.x * 256 + threadIdx.x;
    if (gid < n) p[gid] = 0.0f;
}

__global__ __launch_bounds__(256)
void mpnn_count(const int* __restrict__ mol_index, float* __restrict__ counts) {
    int gid = blockIdx.x * 256 + threadIdx.x;
    if (gid < N_ATOMS) atomicAdd(&counts[mol_index[gid]], 1.0f);
}

__global__ __launch_bounds__(256)
void mpnn_finalize(const float* __restrict__ sums, const float* __restrict__ counts,
                   float* __restrict__ out) {
    int gid = blockIdx.x * 256 + threadIdx.x;
    if (gid >= N_MOLS * HID) return;
    out[gid] = sums[gid] / fmaxf(counts[gid / HID], 1.0f);
}

// ---------------------------------------------------------------------------
// host-side launch
// ---------------------------------------------------------------------------
extern "C" void kernel_launch(void* const* d_in, const int* in_sizes, int n_in,
                              void* d_out, int out_size, void* d_ws, size_t ws_size,
                              hipStream_t stream) {
    (void)in_sizes; (void)n_in; (void)out_size; (void)ws_size;

    const float* f_atoms  = (const float*)d_in[0];
    const float* f_bonds  = (const float*)d_in[1];
    const float* W_i      = (const float*)d_in[2];
    const float* W_m      = (const float*)d_in[3];
    const float* W_a      = (const float*)d_in[4];
    const float* b_a      = (const float*)d_in[5];
    const int*   a2b      = (const int*)d_in[6];
    const int*   b2a      = (const int*)d_in[7];
    const int*   b2revb   = (const int*)d_in[8];
    const int*   mol_idx  = (const int*)d_in[9];
    float*       out      = (float*)d_out;

    // ---- workspace layout (bytes, 256-aligned)
    char* ws = (char*)d_ws;
    size_t off = 0;
    auto take = [&](size_t bytes) { char* p = ws + off; off += (bytes + 255) & ~size_t(255); return p; };

    bf16*  fb16   = (bf16*)take((size_t)N_BONDS * KP_BOND * sizeof(bf16)); // 80 MB
    bf16*  wi_t   = (bf16*)take((size_t)HID * KP_BOND * sizeof(bf16));
    bf16*  wm_t   = (bf16*)take((size_t)HID * HID * sizeof(bf16));
    bf16*  wa_t   = (bf16*)take((size_t)HID * KP_CAT * sizeof(bf16));
    bf16*  inp16  = (bf16*)take((size_t)N_BONDS * HID * sizeof(bf16));     // 128 MB
    bf16*  msg16  = (bf16*)take((size_t)N_BONDS * HID * sizeof(bf16));     // 128 MB
    bf16*  m16    = (bf16*)take((size_t)N_BONDS * HID * sizeof(bf16));     // 128 MB
    float* agg    = (float*)take((size_t)N_ATOMS * HID * sizeof(float));   // 64 MB
    bf16*  ain16  = (bf16*)take((size_t)N_ATOMS * KP_CAT * sizeof(bf16));  // 52 MB
    float* sums   = (float*)take((size_t)N_MOLS * HID * sizeof(float));
    float* counts = (float*)take((size_t)N_MOLS * sizeof(float));

    // ---- 1) precision/layout prepass
    {
        long long n = (long long)N_BONDS * KP_BOND;
        mpnn_conv_fbonds<<<(unsigned)((n + 255) / 256), 256, 0, stream>>>(f_bonds, fb16);
    }
    mpnn_conv_weight_t<<<(HID * KP_BOND + 255) / 256, 256, 0, stream>>>(W_i, BOND_F, HID, KP_BOND, wi_t);
    mpnn_conv_weight_t<<<(HID * HID + 255) / 256, 256, 0, stream>>>(W_m, HID, HID, HID, wm_t);
    mpnn_conv_weight_t<<<(HID * KP_CAT + 255) / 256, 256, 0, stream>>>(W_a, ATOM_F + HID, HID, KP_CAT, wa_t);

    // ---- 2) inp = f_bonds @ W_i ; message = relu(inp)
    {
        dim3 grid(N_BONDS / BM, HID / BN);
        mpnn_gemm_wmma<<<grid, 256, 0, stream>>>(fb16, KP_BOND, wi_t, KP_BOND, KP_BOND,
                                                 /*mode*/0, inp16, msg16,
                                                 nullptr, nullptr, nullptr, nullptr);
    }

    // ---- 3) two message-passing steps
    for (int step = 0; step < 2; ++step) {
        mpnn_atom_agg<<<(N_ATOMS * 32) / 256, 256, 0, stream>>>(msg16, a2b, agg);
        mpnn_bond_m<<<(N_BONDS * 32) / 256, 256, 0, stream>>>(agg, msg16, b2a, b2revb, m16);
        dim3 grid(N_BONDS / BM, HID / BN);
        mpnn_gemm_wmma<<<grid, 256, 0, stream>>>(m16, HID, wm_t, HID, HID,
                                                 /*mode*/1, nullptr, msg16,
                                                 inp16, nullptr, nullptr, nullptr);
    }

    // ---- 4) atom aggregation + concat input
    mpnn_atom_agg<<<(N_ATOMS * 32) / 256, 256, 0, stream>>>(msg16, a2b, agg);
    {
        long long n = (long long)N_ATOMS * KP_CAT;
        mpnn_build_ainput<<<(unsigned)((n + 255) / 256), 256, 0, stream>>>(f_atoms, agg, ain16);
    }

    // ---- 5) segment state, W_a GEMM fused with per-molecule atomic sum
    mpnn_zero<<<(N_MOLS * HID + N_MOLS + 255) / 256, 256, 0, stream>>>(sums, N_MOLS * HID + N_MOLS);
    // counts immediately follows sums in one zero pass? counts is a separate alloc:
    mpnn_zero<<<(N_MOLS + 255) / 256, 256, 0, stream>>>(counts, N_MOLS);
    mpnn_count<<<(N_ATOMS + 255) / 256, 256, 0, stream>>>(mol_idx, counts);
    {
        dim3 grid(N_ATOMS / BM, HID / BN);
        mpnn_gemm_wmma<<<grid, 256, 0, stream>>>(ain16, KP_CAT, wa_t, KP_CAT, KP_CAT,
                                                 /*mode*/2, nullptr, nullptr,
                                                 nullptr, b_a, mol_idx, sums);
    }

    // ---- 6) mean readout
    mpnn_finalize<<<(N_MOLS * HID + 255) / 256, 256, 0, stream>>>(sums, counts, out);
}
